// EquivariantBlock_17179869184408
// MI455X (gfx1250) — matrix operand order repaired and checked by reading, not compiled, not run on hardware
//
#include <hip/hip_runtime.h>
#include <math.h>

// ---------------------------------------------------------------------------
// MI455X / gfx1250 implementation of the EquivariantBlock forward.
// All dense layers run on v_wmma_f32_16x16x32_bf16 (bf16 operands, f32 acc).
// Gathers (hin concat, h_out[row]+h_out[col]) fused into GEMM A-staging.
// Segment reductions via L2-resident float atomics (wave32 everywhere).
// Staging is float4 / ds_store_b64 vectorized; next-tile global_prefetch_b8.
// ---------------------------------------------------------------------------

typedef __attribute__((ext_vector_type(16))) __bf16 v16bf;
typedef __attribute__((ext_vector_type(8)))  __bf16 v8bf;
typedef __attribute__((ext_vector_type(4)))  __bf16 v4bf;
typedef __attribute__((ext_vector_type(8)))  float  v8f;

#define DEV __device__ __forceinline__

// fast silu: x * v_rcp_f32(1 + exp(-x))  (avoids IEEE div expansion)
DEV float siluf(float x) { return x * __builtin_amdgcn_rcpf(1.0f + __expf(-x)); }
DEV float geluf(float x) { return 0.5f * x * (1.0f + erff(x * 0.70710678118f)); }

DEV v4bf to_bf4(float4 f) {
    v4bf r;
    r[0] = (__bf16)f.x; r[1] = (__bf16)f.y; r[2] = (__bf16)f.z; r[3] = (__bf16)f.w;
    return r;
}

// order-preserving float<->uint map for atomicMax on floats
DEV unsigned fenc(float f) {
    unsigned u = __float_as_uint(f);
    return (u & 0x80000000u) ? ~u : (u | 0x80000000u);
}
DEV float fdec(unsigned u) {
    return __uint_as_float((u & 0x80000000u) ? (u & 0x7fffffffu) : ~u);
}

// ---------------------------------------------------------------------------
// Generic WMMA GEMM:  C[M,Nn] = post( A[M,K] @ W[K,Nn] + bias )
// AMODE: 0 plain A; 1 silu(A); 2 A = a0[idx0[m]] + a0[idx1[m]] (lda=K);
//        3 A = concat(a0[idx0[m]](256), a0[idx1[m]](256), a2[m](64), a3[m](64));
//        4 A = concat(a0[m](64), a1[m](64))
// POST:  0 none; 1 gelu; 2 silu; 3 tanh
// ---------------------------------------------------------------------------
struct AArgs {
    const float* a0; const float* a1; const float* a2; const float* a3;
    const int* idx0; const int* idx1;
};

// 4 consecutive K elements (k % 4 == 0; all concat boundaries are multiples of 4)
template <int AMODE>
DEV float4 loadA4(const AArgs& A, int m, int k, int K) {
    if (AMODE == 0) return *(const float4*)&A.a0[m * K + k];
    if (AMODE == 1) {
        float4 f = *(const float4*)&A.a0[m * K + k];
        f.x = siluf(f.x); f.y = siluf(f.y); f.z = siluf(f.z); f.w = siluf(f.w);
        return f;
    }
    if (AMODE == 2) {
        float4 u = *(const float4*)&A.a0[A.idx0[m] * K + k];
        float4 w = *(const float4*)&A.a0[A.idx1[m] * K + k];
        u.x += w.x; u.y += w.y; u.z += w.z; u.w += w.w;
        return u;
    }
    if (AMODE == 3) {
        if (k < 256) return *(const float4*)&A.a0[A.idx0[m] * 256 + k];
        if (k < 512) return *(const float4*)&A.a0[A.idx1[m] * 256 + (k - 256)];
        if (k < 576) return *(const float4*)&A.a2[m * 64 + (k - 512)];
        return *(const float4*)&A.a3[m * 64 + (k - 576)];
    }
    // AMODE 4
    return (k < 64) ? *(const float4*)&A.a0[m * 64 + k]
                    : *(const float4*)&A.a1[m * 64 + (k - 64)];
}

template <int AMODE, int POST>
__global__ __launch_bounds__(256)
void k_gemm(AArgs A, const float* __restrict__ W, const float* __restrict__ bias,
            float* __restrict__ C, int M, int K, int Nn)
{
    // LDS tiles, K-stride padded to 40 halves (80B, 16B aligned, conflict-free)
    __shared__ __bf16 sA[64][40];
    __shared__ __bf16 sB[64][40];   // W tile stored N-major (transposed)

    const int tid  = threadIdx.x;
    const int wave = tid >> 5;
    const int lane = tid & 31;
    const int m0 = blockIdx.y * 64;
    const int n0 = blockIdx.x * 64;
    const int wm = (wave >> 2) * 32;   // wave M offset in tile (0 / 32)
    const int wn = (wave & 3) * 16;    // wave N offset in tile
    const int lrow = lane & 15;
    const int kbA = (lane < 16) ? 0 : 8;    // A frag: K 0-7/16-23 vs 8-15/24-31
    const int kbB = (lane < 16) ? 0 : 8;    // B frag: K 0-7/16-23 vs 8-15/24-31

    v8f acc0 = {};
    v8f acc1 = {};

    for (int k0 = 0; k0 < K; k0 += 32) {
        // ---- stage A tile (64 x 32) as 4-wide vectors, fp32 -> bf16 ----
        #pragma unroll
        for (int i = 0; i < 2; ++i) {
            int item = tid + i * 256;          // 0..511
            int r  = item >> 3;                // 0..63
            int cg = (item & 7) * 4;           // 0,4,...,28
            float4 f = make_float4(0.f, 0.f, 0.f, 0.f);
            int gm = m0 + r;
            if (gm < M) f = loadA4<AMODE>(A, gm, k0 + cg, K);
            *(v4bf*)&sA[r][cg] = to_bf4(f);
        }
        // ---- stage W tile (32 x 64) transposed into sB[n][k], b64 stores ----
        #pragma unroll
        for (int i = 0; i < 2; ++i) {
            int item = tid + i * 256;
            int nn = item & 63;
            int kg = (item >> 6) * 4;          // 0,4,...,28
            const float* wp = &W[(k0 + kg) * Nn + (n0 + nn)];
            float4 f;
            f.x = wp[0]; f.y = wp[Nn]; f.z = wp[2 * Nn]; f.w = wp[3 * Nn];
            *(v4bf*)&sB[nn][kg] = to_bf4(f);
        }
        // ---- prefetch next K tile while this one is consumed ----
        if (k0 + 32 < K) {
            __builtin_prefetch(&W[(k0 + 32 + (tid >> 6) * 4) * Nn + n0 + (tid & 63)], 0, 3);
            if (AMODE == 0 || AMODE == 1) {
                int r = tid >> 3;
                if (m0 + r < M)
                    __builtin_prefetch(&A.a0[(m0 + r) * K + k0 + 32 + (tid & 7) * 4], 0, 3);
            }
        }
        __syncthreads();

        // ---- B fragment (shared by both M sub-tiles) ----
        v8bf b0 = *(const v8bf*)&sB[wn + lrow][kbB];
        v8bf b1 = *(const v8bf*)&sB[wn + lrow][kbB + 16];
        v16bf bf;
        #pragma unroll
        for (int i = 0; i < 8; ++i) { bf[i] = b0[i]; bf[i + 8] = b1[i]; }

        // ---- A fragments + WMMA ----
        {
            v8bf alo = *(const v8bf*)&sA[wm + lrow][kbA];
            v8bf ahi = *(const v8bf*)&sA[wm + lrow][kbA + 16];
            v16bf af;
            #pragma unroll
            for (int i = 0; i < 8; ++i) { af[i] = alo[i]; af[i + 8] = ahi[i]; }
            acc0 = __builtin_amdgcn_wmma_f32_16x16x32_bf16(
                false, af, false, bf, (short)0, acc0, false, false);
        }
        {
            v8bf alo = *(const v8bf*)&sA[wm + 16 + lrow][kbA];
            v8bf ahi = *(const v8bf*)&sA[wm + 16 + lrow][kbA + 16];
            v16bf af;
            #pragma unroll
            for (int i = 0; i < 8; ++i) { af[i] = alo[i]; af[i + 8] = ahi[i]; }
            acc1 = __builtin_amdgcn_wmma_f32_16x16x32_bf16(
                false, af, false, bf, (short)0, acc1, false, false);
        }
        __syncthreads();
    }

    // ---- epilogue: bias + activation + store (C layout: VGPR r <-> M=r/r+8) ----
    const int cn_ = lane & 15;
    const int cmo = (lane < 16) ? 0 : 8;
    const int gn  = n0 + wn + cn_;
    const float bv = bias ? bias[gn] : 0.0f;
    #pragma unroll
    for (int r = 0; r < 8; ++r) {
        float x0 = acc0[r] + bv;
        float x1 = acc1[r] + bv;
        if (POST == 1) { x0 = geluf(x0); x1 = geluf(x1); }
        else if (POST == 2) { x0 = siluf(x0); x1 = siluf(x1); }
        else if (POST == 3) { x0 = tanhf(x0); x1 = tanhf(x1); }
        int gm0 = m0 + wm + r + cmo;
        int gm1 = gm0 + 16;
        if (gm0 < M) C[gm0 * Nn + gn] = x0;
        if (gm1 < M) C[gm1 * Nn + gn] = x1;
    }
}

// ---------------------------------------------------------------------------
// Gaussian basis + CoorsNorm: one wave per edge
// ---------------------------------------------------------------------------
__global__ __launch_bounds__(256)
void k_gbf(const float* __restrict__ pos, const int* __restrict__ row,
           const int* __restrict__ col, const float* __restrict__ means,
           const float* __restrict__ stds, const float* __restrict__ coors_scale,
           float* __restrict__ dist, float* __restrict__ cnb, int E)
{
    int e = blockIdx.x * 8 + (threadIdx.x >> 5);
    if (e >= E) return;
    int lane = threadIdx.x & 31;
    int r = row[e], c = col[e];
    float dx = pos[r * 3 + 0] - pos[c * 3 + 0];
    float dy = pos[r * 3 + 1] - pos[c * 3 + 1];
    float dz = pos[r * 3 + 2] - pos[c * 3 + 2];
    float radial = dx * dx + dy * dy + dz * dz;
    const float inv_s2pi = 0.39894245f;  // 1/sqrt(2*3.14159)
    #pragma unroll
    for (int t = 0; t < 2; ++t) {
        int f = lane + t * 32;
        float val;
        if (f == 0) {
            val = radial;
        } else {
            float sd = fabsf(stds[f - 1]) + 1e-5f;
            float z = (radial - means[f - 1]) * __builtin_amdgcn_rcpf(sd);
            val = __expf(-0.5f * z * z) * inv_s2pi * __builtin_amdgcn_rcpf(sd);
        }
        dist[e * 64 + f] = val;
    }
    if (lane == 0) {
        float nrm = sqrtf(radial);
        float inv = coors_scale[0] * __builtin_amdgcn_rcpf(fmaxf(nrm, 1e-8f));
        cnb[e * 3 + 0] = dx * inv;
        cnb[e * 3 + 1] = dy * inv;
        cnb[e * 3 + 2] = dz * inv;
    }
}

// ---------------------------------------------------------------------------
// out = LN(x) * (1 + t[:, sc_off:]) + t[:, sh_off:]   (wave per row)
// ---------------------------------------------------------------------------
template <int DIM>
__global__ __launch_bounds__(256)
void k_ln_mod(const float* __restrict__ x, const float* __restrict__ t,
              int tstride, int sh_off, int sc_off, float* __restrict__ out, int M)
{
    constexpr int V = DIM / 32;
    int rowi = blockIdx.x * 8 + (threadIdx.x >> 5);
    if (rowi >= M) return;
    int lane = threadIdx.x & 31;
    float v[V];
    if (V == 8) {
        float4 f0 = *(const float4*)&x[rowi * DIM + lane * 8];
        float4 f1 = *(const float4*)&x[rowi * DIM + lane * 8 + 4];
        v[0] = f0.x; v[1] = f0.y; v[2] = f0.z; v[3] = f0.w;
        v[4] = f1.x; v[5] = f1.y; v[6] = f1.z; v[7] = f1.w;
    } else {
        float2 f = *(const float2*)&x[rowi * DIM + lane * 2];
        v[0] = f.x; v[1] = f.y;
    }
    float s = 0.0f, s2 = 0.0f;
    #pragma unroll
    for (int i = 0; i < V; ++i) { s += v[i]; s2 += v[i] * v[i]; }
    #pragma unroll
    for (int m = 16; m; m >>= 1) {
        s  += __shfl_xor(s,  m, 32);
        s2 += __shfl_xor(s2, m, 32);
    }
    float mean = s / DIM;
    float var  = s2 / DIM - mean * mean;
    float rstd = rsqrtf(var + 1e-6f);
    #pragma unroll
    for (int i = 0; i < V; ++i) {
        int j = lane * V + i;
        out[rowi * DIM + j] =
            (v[i] - mean) * rstd * (1.0f + t[rowi * tstride + sc_off + j]) +
            t[rowi * tstride + sh_off + j];
    }
}

// out = a + t[:, g_off:] * b   (gated residual, float4 per thread)
__global__ __launch_bounds__(256)
void k_gate_add(const float* __restrict__ a, const float* __restrict__ b,
                const float* __restrict__ t, int tstride, int g_off,
                float* __restrict__ out, int M, int DIMr)
{
    int idx4 = (blockIdx.x * 256 + threadIdx.x) * 4;
    if (idx4 >= M * DIMr) return;
    int r = idx4 / DIMr, j = idx4 - r * DIMr;   // DIMr multiple of 4 -> same row
    float4 av = *(const float4*)&a[idx4];
    float4 bvv = *(const float4*)&b[idx4];
    float4 gv = *(const float4*)&t[r * tstride + g_off + j];
    float4 o;
    o.x = av.x + gv.x * bvv.x; o.y = av.y + gv.y * bvv.y;
    o.z = av.z + gv.z * bvv.z; o.w = av.w + gv.w * bvv.w;
    *(float4*)&out[idx4] = o;
}

__global__ void k_fillf(float* p, float v, int n) {
    int i = blockIdx.x * 256 + threadIdx.x;
    if (i < n) p[i] = v;
}
__global__ void k_fillu(unsigned* p, unsigned v, int n) {
    int i = blockIdx.x * 256 + threadIdx.x;
    if (i < n) p[i] = v;
}
__global__ void k_copy(const float* s, float* d, int n) {
    int i = blockIdx.x * 256 + threadIdx.x;
    if (i < n) d[i] = s[i];
}

// ---------------------------------------------------------------------------
// Segment softmax attention (3 passes, L2-resident atomics)
// ---------------------------------------------------------------------------
__global__ __launch_bounds__(256)
void k_attn_score(const float* __restrict__ q, const float* __restrict__ kk,
                  const float* __restrict__ ek, const int* __restrict__ row,
                  const int* __restrict__ col, float* __restrict__ alpha,
                  unsigned* __restrict__ amaxU, int E)
{
    int idx = blockIdx.x * 256 + threadIdx.x;
    if (idx >= E * 8) return;
    int e = idx >> 3, h = idx & 7;
    int r = row[e], c = col[e];
    const float* qp = q  + c * 256 + h * 32;
    const float* kp = kk + r * 256 + h * 32;
    const float* ep = ek + e * 256 + h * 32;
    float s = 0.0f;
    #pragma unroll
    for (int i = 0; i < 8; ++i) {
        float4 qv = *(const float4*)(qp + i * 4);
        float4 kv = *(const float4*)(kp + i * 4);
        float4 evv = *(const float4*)(ep + i * 4);
        s += qv.x * kv.x * evv.x + qv.y * kv.y * evv.y +
             qv.z * kv.z * evv.z + qv.w * kv.w * evv.w;
    }
    s *= 0.17677669529f;  // 1/sqrt(32)
    alpha[idx] = s;
    atomicMax(&amaxU[c * 8 + h], fenc(s));
}

__global__ __launch_bounds__(256)
void k_attn_ea(const float* __restrict__ alpha, const unsigned* __restrict__ amaxU,
               const int* __restrict__ col, float* __restrict__ ea,
               float* __restrict__ asum, int E)
{
    int idx = blockIdx.x * 256 + threadIdx.x;
    if (idx >= E * 8) return;
    int e = idx >> 3, h = idx & 7;
    int c = col[e];
    float v = __expf(alpha[idx] - fdec(amaxU[c * 8 + h]));
    ea[idx] = v;
    atomicAdd(&asum[c * 8 + h], v);
}

__global__ __launch_bounds__(256)
void k_attn_msg(const float* __restrict__ v, const float* __restrict__ ev,
                const float* __restrict__ ea, const float* __restrict__ asum,
                const int* __restrict__ row, const int* __restrict__ col,
                float* __restrict__ attn, int E)
{
    int idx = blockIdx.x * 256 + threadIdx.x;
    if (idx >= E * 256) return;
    int e = idx >> 8, j = idx & 255;
    int h = j >> 5;
    int r = row[e], c = col[e];
    float w = ea[e * 8 + h] * __builtin_amdgcn_rcpf(asum[c * 8 + h] + 1e-16f);
    atomicAdd(&attn[c * 256 + j], v[r * 256 + j] * ev[idx] * w);
}

// ---------------------------------------------------------------------------
// pos_out[row[e]] += cn[e] * tanh(dot(c1out[e], w2))   (wave per edge)
// ---------------------------------------------------------------------------
__global__ __launch_bounds__(256)
void k_coord(const float* __restrict__ c1o, const float* __restrict__ w2,
             const float* __restrict__ cnb, const int* __restrict__ row,
             float* __restrict__ pos_out, int E)
{
    int e = blockIdx.x * 8 + (threadIdx.x >> 5);
    if (e >= E) return;
    int lane = threadIdx.x & 31;
    float4 c0 = *(const float4*)&c1o[e * 256 + lane * 8];
    float4 c1 = *(const float4*)&c1o[e * 256 + lane * 8 + 4];
    float4 w0 = *(const float4*)&w2[lane * 8];
    float4 w1 = *(const float4*)&w2[lane * 8 + 4];
    float s = c0.x * w0.x + c0.y * w0.y + c0.z * w0.z + c0.w * w0.w +
              c1.x * w1.x + c1.y * w1.y + c1.z * w1.z + c1.w * w1.w;
    #pragma unroll
    for (int m = 16; m; m >>= 1) s += __shfl_xor(s, m, 32);
    if (lane == 0) {
        float g = tanhf(s);
        int r = row[e];
        atomicAdd(&pos_out[r * 3 + 0], cnb[e * 3 + 0] * g);
        atomicAdd(&pos_out[r * 3 + 1], cnb[e * 3 + 1] * g);
        atomicAdd(&pos_out[r * 3 + 2], cnb[e * 3 + 2] * g);
    }
}

// ---------------------------------------------------------------------------
// Host orchestration
// ---------------------------------------------------------------------------
#define GEMM(AMODE, POST, Aa, Wp, Bp, Cp, Mm, Kk, NN)                         \
    k_gemm<AMODE, POST><<<dim3((NN) / 64, ((Mm) + 63) / 64), 256, 0, stream>>>\
        (Aa, Wp, Bp, Cp, Mm, Kk, NN)

extern "C" void kernel_launch(void* const* d_in, const int* in_sizes, int n_in,
                              void* d_out, int out_size, void* d_ws, size_t ws_size,
                              hipStream_t stream)
{
    const int N = in_sizes[0] / 3;
    const int E = in_sizes[3] / 2;
    const int D = 256, ED = 64, TD = 128;

    const float* pos   = (const float*)d_in[0];
    const float* h     = (const float*)d_in[1];
    const float* eattr = (const float*)d_in[2];
    const int*   eidx  = (const int*)d_in[3];
    const int*   row   = eidx;
    const int*   col   = eidx + E;
    const float* ntime = (const float*)d_in[4];
    const float* etime = (const float*)d_in[5];
    int p = 6;
    const float* gbf_means   = (const float*)d_in[p++];
    const float* gbf_stds    = (const float*)d_in[p++];
    const float* edge_emb_w  = (const float*)d_in[p++];
    const float* edge_emb_b  = (const float*)d_in[p++];
    const float* wq = (const float*)d_in[p++]; const float* bq = (const float*)d_in[p++];
    const float* wk = (const float*)d_in[p++]; const float* bk = (const float*)d_in[p++];
    const float* wv = (const float*)d_in[p++]; const float* bv = (const float*)d_in[p++];
    const float* we0 = (const float*)d_in[p++];
    const float* we1 = (const float*)d_in[p++];
    const float* proj_w = (const float*)d_in[p++]; const float* proj_b = (const float*)d_in[p++];
    const float* ff1_w = (const float*)d_in[p++];  const float* ff1_b = (const float*)d_in[p++];
    const float* ff2_w = (const float*)d_in[p++];  const float* ff2_b = (const float*)d_in[p++];
    const float* n2e_w = (const float*)d_in[p++];  const float* n2e_b = (const float*)d_in[p++];
    const float* ff3_w = (const float*)d_in[p++];  const float* ff3_b = (const float*)d_in[p++];
    const float* ff4_w = (const float*)d_in[p++];  const float* ff4_b = (const float*)d_in[p++];
    const float* node_time_w = (const float*)d_in[p++]; const float* node_time_b = (const float*)d_in[p++];
    const float* edge_time_w = (const float*)d_in[p++]; const float* edge_time_b = (const float*)d_in[p++];
    const float* coors_scale = (const float*)d_in[p++];
    const float* eq_time_w = (const float*)d_in[p++]; const float* eq_time_b = (const float*)d_in[p++];
    const float* eq_in_w   = (const float*)d_in[p++]; const float* eq_in_b   = (const float*)d_in[p++];
    const float* eq_c1_w   = (const float*)d_in[p++]; const float* eq_c1_b   = (const float*)d_in[p++];
    const float* eq_c2_w   = (const float*)d_in[p++];

    // output regions (tuple: h_out, edge_out, pos_out concatenated flat)
    float* h_out    = (float*)d_out;
    float* edge_out = h_out + (size_t)N * D;
    float* pos_out  = edge_out + (size_t)E * ED;

    // workspace carve
    char* wsp = (char*)d_ws;
    auto alloc = [&](size_t elems) -> float* {
        float* r = (float*)wsp;
        wsp += ((elems * 4 + 255) / 256) * 256;
        return r;
    };
    float* dist  = alloc((size_t)E * 64);
    float* cnb   = alloc((size_t)E * 3);
    float* tn    = alloc((size_t)N * 1536);
    float* te    = alloc((size_t)E * 384);
    float* ebuf  = alloc((size_t)E * 64);
    float* hmod  = alloc((size_t)N * 256);
    float* emod  = alloc((size_t)E * 64);
    float* qb    = alloc((size_t)N * 256);
    float* kbuf  = alloc((size_t)N * 256);
    float* vbuf  = alloc((size_t)N * 256);
    float* ekb   = alloc((size_t)E * 256);
    float* evb   = alloc((size_t)E * 256);
    float* alpha = alloc((size_t)E * 8);
    float* eab   = alloc((size_t)E * 8);
    unsigned* amaxU = (unsigned*)alloc((size_t)N * 8);
    float* asum  = alloc((size_t)N * 8);
    float* attn  = alloc((size_t)N * 256);
    float* attnp = alloc((size_t)N * 256);
    float* hnode = alloc((size_t)N * 256);
    float* hf    = alloc((size_t)N * 256);
    float* ffmid = alloc((size_t)N * 1024);
    float* ff2o  = alloc((size_t)N * 256);
    float* enode = alloc((size_t)E * 64);
    float* hedge = alloc((size_t)E * 64);
    float* efb   = alloc((size_t)E * 64);
    float* ffe   = alloc((size_t)E * 256);
    float* ff4o  = alloc((size_t)E * 64);
    float* tqb   = alloc((size_t)E * 512);
    float* inv0  = alloc((size_t)E * 256);
    float* c1o   = alloc((size_t)E * 256);

    auto A0 = [&](const float* a) { AArgs x{}; x.a0 = a; return x; };

    // 1) Gaussian basis + CoorsNorm
    k_gbf<<<(E + 7) / 8, 256, 0, stream>>>(pos, row, col, gbf_means, gbf_stds,
                                           coors_scale, dist, cnb, E);
    // 2) time MLPs: silu(t) @ W + b
    GEMM(1, 0, A0(ntime), node_time_w, node_time_b, tn, N, TD, 6 * D);
    GEMM(1, 0, A0(etime), edge_time_w, edge_time_b, te, E, TD, 6 * ED);
    // 3) edge embedding: concat(dist, edge_attr) @ W + b
    { AArgs a{}; a.a0 = dist; a.a1 = eattr;
      GEMM(4, 0, a, edge_emb_w, edge_emb_b, ebuf, E, 2 * ED, ED); }
    // 4) adaLN modulation
    k_ln_mod<256><<<(N + 7) / 8, 256, 0, stream>>>(h, tn, 6 * D, 0, D, hmod, N);
    k_ln_mod<64><<<(E + 7) / 8, 256, 0, stream>>>(ebuf, te, 6 * ED, 0, ED, emod, E);
    // 5) Q,K,V + edge gates
    GEMM(0, 0, A0(hmod), wq, bq, qb, N, D, D);
    GEMM(0, 0, A0(hmod), wk, bk, kbuf, N, D, D);
    GEMM(0, 0, A0(hmod), wv, bv, vbuf, N, D, D);
    GEMM(0, 3, A0(emod), we0, (const float*)nullptr, ekb, E, ED, D);  // tanh
    GEMM(0, 3, A0(emod), we1, (const float*)nullptr, evb, E, ED, D);  // tanh
    // 6) segment softmax attention
    k_fillu<<<(N * 8 + 255) / 256, 256, 0, stream>>>(amaxU, 0u, N * 8);
    k_fillf<<<(N * 8 + 255) / 256, 256, 0, stream>>>(asum, 0.0f, N * 8);
    k_fillf<<<(N * 256 + 255) / 256, 256, 0, stream>>>(attn, 0.0f, N * 256);
    k_attn_score<<<(E * 8 + 255) / 256, 256, 0, stream>>>(qb, kbuf, ekb, row, col,
                                                          alpha, amaxU, E);
    k_attn_ea<<<(E * 8 + 255) / 256, 256, 0, stream>>>(alpha, amaxU, col, eab, asum, E);
    k_attn_msg<<<(E * 256 + 255) / 256, 256, 0, stream>>>(vbuf, evb, eab, asum,
                                                          row, col, attn, E);
    // 7) output projection + gated residual
    GEMM(0, 0, A0(attn), proj_w, proj_b, attnp, N, D, D);
    k_gate_add<<<(N * 256 / 4 + 255) / 256, 256, 0, stream>>>(h, attnp, tn, 6 * D,
                                                              2 * D, hnode, N, 256);
    // 8) node FFN
    k_ln_mod<256><<<(N + 7) / 8, 256, 0, stream>>>(hnode, tn, 6 * D, 3 * D, 4 * D, hf, N);
    GEMM(0, 1, A0(hf), ff1_w, ff1_b, ffmid, N, D, 4 * D);        // gelu
    GEMM(0, 0, A0(ffmid), ff2_w, ff2_b, ff2o, N, 4 * D, D);
    k_gate_add<<<(N * 256 / 4 + 255) / 256, 256, 0, stream>>>(hnode, ff2o, tn, 6 * D,
                                                              5 * D, h_out, N, 256);
    // 9) edge update: (h_out[row]+h_out[col]) @ n2e  (gather fused into GEMM)
    { AArgs a{}; a.a0 = h_out; a.idx0 = row; a.idx1 = col;
      GEMM(2, 0, a, n2e_w, n2e_b, enode, E, D, ED); }
    k_gate_add<<<(E * 64 / 4 + 255) / 256, 256, 0, stream>>>(eattr, enode, te, 6 * ED,
                                                             2 * ED, hedge, E, 64);
    // 10) edge FFN
    k_ln_mod<64><<<(E + 7) / 8, 256, 0, stream>>>(hedge, te, 6 * ED, 3 * ED, 4 * ED, efb, E);
    GEMM(0, 1, A0(efb), ff3_w, ff3_b, ffe, E, ED, 4 * ED);       // gelu
    GEMM(0, 0, A0(ffe), ff4_w, ff4_b, ff4o, E, 4 * ED, ED);
    k_gate_add<<<(E * 64 / 4 + 255) / 256, 256, 0, stream>>>(hedge, ff4o, te, 6 * ED,
                                                             5 * ED, edge_out, E, 64);
    // 11) equivariant coordinate update
    GEMM(1, 0, A0(etime), eq_time_w, eq_time_b, tqb, E, TD, 2 * D);   // silu preact
    { AArgs a{}; a.a0 = h_out; a.idx0 = row; a.idx1 = col; a.a2 = edge_out; a.a3 = dist;
      GEMM(3, 0, a, eq_in_w, eq_in_b, inv0, E, 2 * D + 2 * ED, D); }  // hin fused
    k_ln_mod<256><<<(E + 7) / 8, 256, 0, stream>>>(inv0, tqb, 2 * D, 0, D, inv0, E);
    GEMM(0, 2, A0(inv0), eq_c1_w, eq_c1_b, c1o, E, D, D);             // silu post
    k_copy<<<(N * 3 + 255) / 256, 256, 0, stream>>>(pos, pos_out, N * 3);
    k_coord<<<(E + 7) / 8, 256, 0, stream>>>(c1o, eq_c2_w, cnb, row, pos_out, E);
    (void)n_in; (void)out_size; (void)ws_size;
}
